// MultiHeadAttention_66408784331203
// MI455X (gfx1250) — matrix operand order repaired and checked by reading, not compiled, not run on hardware
//
#include <hip/hip_runtime.h>

typedef __attribute__((ext_vector_type(16))) __bf16        bf16x16;
typedef __attribute__((ext_vector_type(8)))  float         f32x8;
typedef __attribute__((ext_vector_type(4)))  unsigned int  u32x4;
typedef unsigned short u16;
typedef unsigned int   u32;

static constexpr int B_  = 2;
static constexpr int S_  = 2048;
static constexpr int D_  = 1024;
static constexpr int H_  = 16;
static constexpr int HD_ = 64;

#define LOG2E_F 1.4426950408889634f

// ---------------------------------------------------------------- helpers --
__device__ __forceinline__ u16 f2bf(float f) {
    union { __bf16 h; u16 u; } c;
    c.h = (__bf16)f;  // RNE float->bf16
    return c.u;
}

union FragU { bf16x16 v; u32x4 q[2]; };

// A fragment (16x32 bf16): per-lane row is K-contiguous; elements 0..15 at
// K = kc + half*16 + s.
__device__ __forceinline__ bf16x16 load_afrag(const u16* __restrict__ rowptr,
                                              int kc, int half) {
    FragU u;
    u.q[0] = *(const u32x4*)(rowptr + kc + half * 16);
    u.q[1] = *(const u32x4*)(rowptr + kc + half * 16 + 8);
    return u.v;
}

// B fragment: 16 contraction-contiguous bf16 (caller pre-offsets by half*16).
__device__ __forceinline__ bf16x16 load_bfrag_lds(const u16* p) {
    FragU u;
    u.q[0] = *(const u32x4*)(p);
    u.q[1] = *(const u32x4*)(p + 8);
    return u.v;
}

// CDNA5 async copy: one b128 per lane, global -> LDS, tracked by ASYNCcnt.
__device__ __forceinline__ void async_b128(u16* lds_dst, const u16* gsrc) {
    u32 l = (u32)(uintptr_t)lds_dst;  // low 32 bits of generic ptr = LDS offset
    asm volatile("global_load_async_to_lds_b128 %0, %1, off"
                 :: "v"(l), "v"(gsrc)
                 : "memory");
}
__device__ __forceinline__ void wait_async_le1() {
    asm volatile("s_wait_asynccnt 0x1" ::: "memory");
}
__device__ __forceinline__ void wait_async_le4() {
    asm volatile("s_wait_asynccnt 0x4" ::: "memory");
}
__device__ __forceinline__ void wait_async_0() {
    asm volatile("s_wait_asynccnt 0x0" ::: "memory");
}

// -------------------------------------------------------- f32 -> bf16 bits --
__global__ void k_f32_to_bf16(const float* __restrict__ in,
                              u16* __restrict__ out, int n) {
    int i      = blockIdx.x * blockDim.x + threadIdx.x;
    int stride = gridDim.x * blockDim.x;
    for (; i < n; i += stride) out[i] = f2bf(in[i]);
}

// ------------------------------------------------------------------- GEMM --
// C[M,N] = A[M,K](bf16,row-major) @ W[N,K](bf16,row-major)^T + bias[N]
// mode 0: bf16 [M,N] | mode 1: bf16 V^T [B,H,HD,S] | mode 2: f32 [M,N]
// Block: 256 threads = 8 waves; tile M=128, N=64.
// W tile (64x32) staged in LDS via async copy; pipeline unrolled by 2 with a
// peeled tail so buffer indices are compile-time constants (no branches, no
// fragment register copies in steady state). Requires K % 64 == 0, K >= 128.
__global__ __launch_bounds__(256)
void k_gemm_bf16(const u16* __restrict__ A, const u16* __restrict__ W,
                 const float* __restrict__ bias, void* __restrict__ outp,
                 int M, int N, int K, int mode) {
    // padded to 40 elems (80B): b128 reads are LDS-bank conflict free
    __shared__ __attribute__((aligned(16))) u16 wtile[2][64][40];

    const int lane  = threadIdx.x & 31;
    const int wave  = threadIdx.x >> 5;
    const int half  = lane >> 4;
    const int l16   = lane & 15;
    const int mrow  = blockIdx.x * 128 + wave * 16;
    const int nbase = blockIdx.y * 64;

    // async-copy mapping: 64 rows x 4 chunks(16B) = 256 chunks, 1 per thread
    const int  crow = threadIdx.x >> 2;
    const int  ccol = threadIdx.x & 3;
    const u16* wsrc = W + (size_t)(nbase + crow) * K + ccol * 8;

    f32x8      acc[4] = {};
    const u16* arow   = A + (size_t)(mrow + l16) * K;

    auto stage = [&](int bi, int kk) {
        async_b128(&wtile[bi][crow][ccol * 8], wsrc + kk);
    };
    auto compute = [&](int bi, bf16x16 af) {
        __syncthreads();  // all waves' tile copies visible
#pragma unroll
        for (int t = 0; t < 4; ++t) {
            bf16x16 bf = load_bfrag_lds(&wtile[bi][t * 16 + l16][half * 16]);
            acc[t]     = __builtin_amdgcn_wmma_f32_16x16x32_bf16(
                false, af, false, bf, (short)0, acc[t], false, false);
        }
        __syncthreads();  // tile fully read before it is restaged
    };

    stage(0, 0);
    bf16x16 af0 = load_afrag(arow, 0, half);
    stage(1, 32);
    bf16x16 af1 = load_afrag(arow, 32, half);

    int kk = 0;
    for (; kk < K - 64; kk += 64) {
        wait_async_le1();  // oldest (buffer 0) staged
        compute(0, af0);
        stage(0, kk + 64);
        af0 = load_afrag(arow, kk + 64, half);

        wait_async_le1();  // buffer 1 staged
        compute(1, af1);
        stage(1, kk + 96);
        af1 = load_afrag(arow, kk + 96, half);
    }
    wait_async_le1();
    compute(0, af0);
    wait_async_0();
    compute(1, af1);

#pragma unroll
    for (int t = 0; t < 4; ++t) {
        const int   ng = nbase + t * 16 + l16;
        const float bv = bias[ng];
#pragma unroll
        for (int r = 0; r < 8; ++r) {
            const int   mg = mrow + r + 8 * half;
            const float v  = acc[t][r] + bv;
            if (mode == 2) {
                ((float*)outp)[(size_t)mg * N + ng] = v;
            } else if (mode == 0) {
                ((u16*)outp)[(size_t)mg * N + ng] = f2bf(v);
            } else {  // V^T: [B,H,HD,S]
                const int hh = ng >> 6, dd = ng & 63;
                const int bb = mg / S_, ss = mg % S_;
                ((u16*)outp)[(((size_t)bb * H_ + hh) * HD_ + dd) * S_ + ss] =
                    f2bf(v);
            }
        }
    }
}

// -------------------------------------------------------- flash attention --
// Q,K: bf16 [B,S,D]; Vt: bf16 [B,H,HD,S]; O: bf16 [B,S,D]. Causal, scale=1/8.
// Block = 128 threads = 4 waves; block covers 64 query rows (16/wave).
// 32-key K/V tiles staged in LDS via async copy (shared by all 4 waves);
// pipeline unrolled by 2 (numkb = 2*(blockIdx.x+1) is always even).
__global__ __launch_bounds__(128)
void k_flash(const u16* __restrict__ Q, const u16* __restrict__ Kb,
             const u16* __restrict__ Vt, u16* __restrict__ O) {
    __shared__ __attribute__((aligned(16))) u16 ktile[2][32][72];  // [key][hd]
    __shared__ __attribute__((aligned(16))) u16 vtile[2][64][40];  // [hd][key]
    __shared__ __attribute__((aligned(16))) u16 plds[4][16][40];

    const int   lane  = threadIdx.x & 31;
    const int   wave  = threadIdx.x >> 5;
    const int   half  = lane >> 4;
    const int   l16   = lane & 15;
    const int   b     = blockIdx.y / H_;
    const int   h     = blockIdx.y % H_;
    const int   q0    = blockIdx.x * 64;
    const int   qrow  = q0 + wave * 16;
    const float scale = 0.125f;  // 1/sqrt(64)

    // async-copy mappings (128 threads):
    const int kc0row = threadIdx.x >> 3;  // K tile: 32 rows x 8 chunks(16B)
    const int kc0col = threadIdx.x & 7;
    const int vc0row = threadIdx.x >> 2;  // V tile: 64 rows x 4 chunks(16B)
    const int vc0col = threadIdx.x & 3;

    const u16* ksrc = Kb + ((size_t)b * S_) * D_ + h * HD_;
    const u16* vsrc = Vt + ((size_t)(b * H_ + h) * HD_) * S_;

    // Q fragments resident in registers for the whole key loop (hd=64 -> 2).
    const u16* qp  = Q + ((size_t)b * S_ + qrow + l16) * D_ + h * HD_;
    bf16x16    qa0 = load_afrag(qp, 0, half);
    bf16x16    qa1 = load_afrag(qp, 32, half);

    f32x8 acc[4] = {};
    float mst[8], lst[8];
#pragma unroll
    for (int r = 0; r < 8; ++r) { mst[r] = -__builtin_inff(); lst[r] = 0.f; }

    const int numkb = q0 / 32 + 2;  // even, >= 2

    auto stageKV = [&](int bi, int kbidx) {
        const int nb = kbidx * 32;
        async_b128(&ktile[bi][kc0row][kc0col * 8],
                   ksrc + (size_t)(nb + kc0row) * D_ + kc0col * 8);
        async_b128(&ktile[bi][kc0row + 16][kc0col * 8],
                   ksrc + (size_t)(nb + kc0row + 16) * D_ + kc0col * 8);
        async_b128(&vtile[bi][vc0row][vc0col * 8],
                   vsrc + (size_t)vc0row * S_ + nb + vc0col * 8);
        async_b128(&vtile[bi][vc0row + 32][vc0col * 8],
                   vsrc + (size_t)(vc0row + 32) * S_ + nb + vc0col * 8);
    };

    auto process = [&](int bi, int kbase) {
        __syncthreads();  // current K/V tiles fully staged (all waves)

        // ---- scores: two 16x16 tiles over this 32-key block ----------------
        float sv[2][8];
#pragma unroll
        for (int n16 = 0; n16 < 2; ++n16) {
            f32x8 st = {};
            st       = __builtin_amdgcn_wmma_f32_16x16x32_bf16(
                false, qa0, false,
                load_bfrag_lds(&ktile[bi][n16 * 16 + l16][half * 16]),
                (short)0, st, false, false);
            st = __builtin_amdgcn_wmma_f32_16x16x32_bf16(
                false, qa1, false,
                load_bfrag_lds(&ktile[bi][n16 * 16 + l16][32 + half * 16]),
                (short)0, st, false, false);
            const int key = kbase + n16 * 16 + l16;
#pragma unroll
            for (int r = 0; r < 8; ++r) {
                const int qg = qrow + r + 8 * half;
                float     s  = st[r] * scale;
                if (key > qg) s = -__builtin_inff();
                sv[n16][r] = s;
            }
        }

        // ---- online softmax: row max / sum via 16-lane shfl reductions -----
        float al[8];
#pragma unroll
        for (int r = 0; r < 8; ++r) {
            float bm = fmaxf(sv[0][r], sv[1][r]);
            bm       = fmaxf(bm, __shfl_xor(bm, 1));
            bm       = fmaxf(bm, __shfl_xor(bm, 2));
            bm       = fmaxf(bm, __shfl_xor(bm, 4));
            bm       = fmaxf(bm, __shfl_xor(bm, 8));
            const float mn = fmaxf(mst[r], bm);
            al[r]          = exp2f((mst[r] - mn) * LOG2E_F);
            mst[r]         = mn;
            const float p0 = exp2f((sv[0][r] - mn) * LOG2E_F);
            const float p1 = exp2f((sv[1][r] - mn) * LOG2E_F);
            sv[0][r]       = p0;
            sv[1][r]       = p1;
            float rs       = p0 + p1;
            rs += __shfl_xor(rs, 1);
            rs += __shfl_xor(rs, 2);
            rs += __shfl_xor(rs, 4);
            rs += __shfl_xor(rs, 8);
            lst[r] = lst[r] * al[r] + rs;
        }
#pragma unroll
        for (int t = 0; t < 4; ++t)
#pragma unroll
            for (int r = 0; r < 8; ++r) acc[t][r] *= al[r];

        // ---- P: C-layout -> A-layout via per-wave LDS tile -----------------
#pragma unroll
        for (int n16 = 0; n16 < 2; ++n16)
#pragma unroll
            for (int r = 0; r < 8; ++r)
                plds[wave][r + 8 * half][n16 * 16 + l16] = f2bf(sv[n16][r]);
        __syncthreads();

        FragU pa;
        pa.q[0] = *(const u32x4*)&plds[wave][l16][half * 16];
        pa.q[1] = *(const u32x4*)&plds[wave][l16][half * 16 + 8];

        // ---- PV: 4 WMMAs against staged V^T tile ---------------------------
#pragma unroll
        for (int t = 0; t < 4; ++t) {
            bf16x16 vf = load_bfrag_lds(&vtile[bi][t * 16 + l16][half * 16]);
            acc[t]     = __builtin_amdgcn_wmma_f32_16x16x32_bf16(
                false, pa.v, false, vf, (short)0, acc[t], false, false);
        }
        __syncthreads();  // tiles fully read before this buffer is restaged
    };

    stageKV(0, 0);
    stageKV(1, 1);

    int kb = 0;
    for (; kb < numkb - 2; kb += 2) {
        wait_async_le4();  // oldest group (buffer 0) staged
        process(0, kb * 32);
        stageKV(0, kb + 2);

        wait_async_le4();  // buffer 1 staged
        process(1, (kb + 1) * 32);
        stageKV(1, kb + 3);
    }
    wait_async_le4();
    process(0, kb * 32);
    wait_async_0();
    process(1, (kb + 1) * 32);

    // ---- finalize: divide by l, write bf16 [B,S,D] -------------------------
#pragma unroll
    for (int r = 0; r < 8; ++r) lst[r] = 1.f / lst[r];
#pragma unroll
    for (int t = 0; t < 4; ++t)
#pragma unroll
        for (int r = 0; r < 8; ++r)
            O[((size_t)b * S_ + qrow + r + 8 * half) * D_ + h * HD_ + t * 16 +
              l16] = f2bf(acc[t][r] * lst[r]);
}

// ---------------------------------------------------------------- launch --
extern "C" void kernel_launch(void* const* d_in, const int* in_sizes, int n_in,
                              void* d_out, int out_size, void* d_ws,
                              size_t ws_size, hipStream_t stream) {
    (void)in_sizes; (void)n_in; (void)out_size; (void)ws_size;

    const float* x  = (const float*)d_in[0];
    const float* Wq = (const float*)d_in[1];
    const float* bq = (const float*)d_in[2];
    const float* Wk = (const float*)d_in[3];
    const float* bk = (const float*)d_in[4];
    const float* Wv = (const float*)d_in[5];
    const float* bv = (const float*)d_in[6];
    const float* Wo = (const float*)d_in[7];
    const float* bo = (const float*)d_in[8];

    const size_t NX = (size_t)B_ * S_ * D_;  // 4,194,304
    const size_t NW = (size_t)D_ * D_;       // 1,048,576

    u16* ws  = (u16*)d_ws;
    u16* xb  = ws;
    u16* wqb = xb + NX;
    u16* wkb = wqb + NW;
    u16* wvb = wkb + NW;
    u16* wob = wvb + NW;
    u16* qb  = wob + NW;
    u16* kbb = qb + NX;
    u16* vtb = kbb + NX;
    u16* ab  = vtb + NX;

    k_f32_to_bf16<<<1024, 256, 0, stream>>>(x, xb, (int)NX);
    k_f32_to_bf16<<<256, 256, 0, stream>>>(Wq, wqb, (int)NW);
    k_f32_to_bf16<<<256, 256, 0, stream>>>(Wk, wkb, (int)NW);
    k_f32_to_bf16<<<256, 256, 0, stream>>>(Wv, wvb, (int)NW);
    k_f32_to_bf16<<<256, 256, 0, stream>>>(Wo, wob, (int)NW);

    const int M = B_ * S_;      // 4096
    dim3 gg(M / 128, D_ / 64);  // (32, 16)
    dim3 gb(256);
    k_gemm_bf16<<<gg, gb, 0, stream>>>(xb, wqb, bq, qb, M, D_, D_, 0);
    k_gemm_bf16<<<gg, gb, 0, stream>>>(xb, wkb, bk, kbb, M, D_, D_, 0);
    k_gemm_bf16<<<gg, gb, 0, stream>>>(xb, wvb, bv, vtb, M, D_, D_, 1);

    dim3 ag(S_ / 64, B_ * H_);  // (32, 32)
    k_flash<<<ag, 128, 0, stream>>>(qb, kbb, vtb, ab);

    k_gemm_bf16<<<gg, gb, 0, stream>>>(ab, wob, bo, (float*)d_out, M, D_, D_, 2);
}
